// MultiHeadAttentionLayer_25314537242917
// MI455X (gfx1250) — matrix operand order repaired and checked by reading, not compiled, not run on hardware
//
#include <hip/hip_runtime.h>
#include <hip/hip_bf16.h>

typedef _Float16 h16;
typedef __attribute__((ext_vector_type(16))) _Float16 v16h;
typedef __attribute__((ext_vector_type(8)))  _Float16 v8h;
typedef __attribute__((ext_vector_type(8)))  float    v8f;

#define HID 256
#define FFNW 512

// f16 weight-fragment region offsets (in halves) inside d_ws
constexpr size_t OFF_W1  = 0;        // 512x256
constexpr size_t OFF_W2  = 131072;   // 256x512
constexpr size_t OFF_WC  = 262144;   // 256x256
constexpr size_t OFF_WQ  = 327680;
constexpr size_t OFF_WK  = 393216;
constexpr size_t OFF_WV  = 458752;
constexpr size_t OFF_WFC = 524288;
constexpr size_t OFF_WZ  = 589824;
constexpr size_t OFF_UZ  = 655360;
constexpr size_t OFF_WR  = 720896;
constexpr size_t OFF_UR  = 786432;
constexpr size_t OFF_WG  = 851968;
constexpr size_t OFF_UG  = 917504;
constexpr size_t OFF_WF  = 983040;
constexpr size_t W_HALVES = 1048576; // total halves (2 MB)

__device__ __forceinline__ float sigm(float x){ return 1.0f / (1.0f + __expf(-x)); }

__device__ __forceinline__ v8f zero8(){
  v8f z = {0.f,0.f,0.f,0.f,0.f,0.f,0.f,0.f};
  return z;
}

__device__ __forceinline__ v8f wmma16(v16h a, v16h b, v8f c){
  // D = A(16x32 f16) * B(32x16 f16) + C(16x16 f32)
  return __builtin_amdgcn_wmma_f32_16x16x32_f16(false, a, false, b, (short)0, c, false, false);
}

// A fragment from f16 LDS strip (row-major, stride S halves), rows 0..15, K block k0..k0+31.
// ISA 16-bit A layout: lane l<16 -> row l, K {k0..k0+7, k0+16..k0+23};
//                      lane l+16 -> row l, K {k0+8..k0+15, k0+24..k0+31}.
__device__ __forceinline__ v16h load_a(const h16* strip, int S, int k0, int lane){
  int r = lane & 15;
  int k = k0 + ((lane & 16) ? 8 : 0);
  const h16* p = strip + r * S + k;
  v8h lo = *(const v8h*)(p);
  v8h hi = *(const v8h*)(p + 16);
  v16h a;
  #pragma unroll
  for (int i = 0; i < 8; ++i){ a[i] = lo[i]; a[i+8] = hi[i]; }
  return a;
}

// B fragment: pre-swizzled so each lane reads 16 contiguous halves (32B), coalesced.
__device__ __forceinline__ v16h load_b(const h16* wf, int frag, int lane){
  return *(const v16h*)(wf + ((size_t)frag * 32 + (size_t)lane) * 16);
}

template<int KB>
__device__ __forceinline__ void load_A(v16h* A, const h16* strip, int S, int lane){
  #pragma unroll
  for (int kb = 0; kb < KB; ++kb) A[kb] = load_a(strip, S, kb * 32, lane);
}

template<int KB>
__device__ __forceinline__ v8f gemm_cb(const v16h* A, const h16* wf, int cb, int lane, v8f acc){
  #pragma unroll
  for (int kb = 0; kb < KB; ++kb) acc = wmma16(A[kb], load_b(wf, cb * KB + kb, lane), acc);
  return acc;
}

// ---------------- weight prep: f32 [O x K] row-major -> swizzled f16 B fragments ----------
__global__ __launch_bounds__(32) void prep_w(const float* __restrict__ src,
                                             h16* __restrict__ dst, int O, int K){
  int frag = blockIdx.x;          // cb * (K/32) + kb
  int l    = threadIdx.x;         // 0..31
  int KB   = K >> 5;
  int cb   = frag / KB;
  int kb   = frag - cb * KB;
  int row  = cb * 16 + (l & 15);                 // W row == B column
  int k    = kb * 32 + ((l & 16) ? 8 : 0);       // lane<16: K {0..7,16..23}, lane>=16: K {8..15,24..31}
  h16* d = dst + ((size_t)frag * 32 + (size_t)l) * 16;
  const float* s = src + (size_t)row * K + k;
  #pragma unroll
  for (int i = 0; i < 16; ++i) d[i] = (h16)s[i < 8 ? i : i + 8];
}

// ---------------- stage1: t=relu(hW1+b1); h2=(tW2+b2)*sig(h); hc_pre + BN partial sums ----
__global__ __launch_bounds__(32) void stage1(const float* __restrict__ h,
                                             const float* __restrict__ b1,
                                             const float* __restrict__ b2,
                                             const float* __restrict__ bc,
                                             const float* __restrict__ wd,
                                             const float* __restrict__ bd,
                                             const h16*  __restrict__ wf,
                                             float* __restrict__ g_h2,
                                             float* __restrict__ partials){
  __shared__ h16 sA[16 * HID];    // h (f16)
  __shared__ h16 sT[16 * FFNW];   // t
  __shared__ h16 s2[16 * HID];    // h2
  int lane = threadIdx.x;
  long long row0 = (long long)blockIdx.x * 16;
  int cL = lane & 15;
  int rb = (lane & 16) ? 8 : 0;

  // stage h -> LDS f16
  for (int idx = lane; idx < 16 * 64; idx += 32){
    int r = idx >> 6, q4 = idx & 63;
    float4 f = ((const float4*)(h + (row0 + r) * HID))[q4];
    h16* p = sA + r * HID + q4 * 4;
    p[0]=(h16)f.x; p[1]=(h16)f.y; p[2]=(h16)f.z; p[3]=(h16)f.w;
  }
  __syncthreads();

  // GEMM1: t = relu(h @ W1^T + b1)   K=256, O=512
  {
    v16h A[8]; load_A<8>(A, sA, HID, lane);
    for (int cb = 0; cb < 32; ++cb){
      v8f acc = gemm_cb<8>(A, wf + OFF_W1, cb, lane, zero8());
      int c = cb * 16 + cL;
      float bb = b1[c];
      #pragma unroll
      for (int v = 0; v < 8; ++v){
        float x = acc[v] + bb;
        x = x > 0.f ? x : 0.f;
        sT[(rb + v) * FFNW + c] = (h16)x;
      }
    }
  }
  __syncthreads();

  // GEMM2: h2 = (t @ W2^T + b2) * sigmoid(h)   K=512, O=256; spill f32 + keep f16
  {
    v16h A[16]; load_A<16>(A, sT, FFNW, lane);
    for (int cb = 0; cb < 16; ++cb){
      v8f acc = gemm_cb<16>(A, wf + OFF_W2, cb, lane, zero8());
      int c = cb * 16 + cL;
      float bb = b2[c];
      #pragma unroll
      for (int v = 0; v < 8; ++v){
        int r = rb + v;
        float x = (acc[v] + bb) * sigm((float)sA[r * HID + c]);
        g_h2[(row0 + r) * HID + c] = x;
        s2[r * HID + c] = (h16)x;
      }
    }
  }
  __syncthreads();

  // GEMM3: hc_pre = (h2 @ Wc^T + bc)*sigmoid(h2)*wd + bd; column partial sum / sumsq
  {
    v16h A[8]; load_A<8>(A, s2, HID, lane);
    for (int cb = 0; cb < 16; ++cb){
      v8f acc = gemm_cb<8>(A, wf + OFF_WC, cb, lane, zero8());
      int c = cb * 16 + cL;
      float bb = bc[c], w = wd[c], b = bd[c];
      float s = 0.f, sq = 0.f;
      #pragma unroll
      for (int v = 0; v < 8; ++v){
        int r = rb + v;
        float x = (acc[v] + bb) * sigm((float)s2[r * HID + c]);
        x = x * w + b;
        s += x; sq += x * x;
      }
      // lanes l and l^16 hold the same column (rows 0-7 / 8-15): deterministic combine
      s  += __shfl_xor(s, 16);
      sq += __shfl_xor(sq, 16);
      if (lane < 16){
        partials[(size_t)blockIdx.x * 512 + c]       = s;
        partials[(size_t)blockIdx.x * 512 + 256 + c] = sq;
      }
    }
  }
}

// ---------------- BN reduce (two deterministic stages) ------------------------------------
__global__ __launch_bounds__(512) void bn_reduce1(const float* __restrict__ partials,
                                                  float* __restrict__ partials2,
                                                  int nb, int chunk){
  int j = threadIdx.x;     // 0..511 (sum | sumsq interleaved blocks)
  int b = blockIdx.x;
  float s = 0.f;
  for (int i = 0; i < chunk; ++i){
    int blk = b * chunk + i;
    if (blk < nb) s += partials[(size_t)blk * 512 + j];
  }
  partials2[(size_t)b * 512 + j] = s;
}

__global__ __launch_bounds__(512) void bn_reduce2(const float* __restrict__ partials2,
                                                  const float* __restrict__ gamma,
                                                  const float* __restrict__ beta,
                                                  float* __restrict__ bn,
                                                  int nb2, float invN){
  __shared__ float tot[512];
  int j = threadIdx.x;
  float s = 0.f;
  for (int i = 0; i < nb2; ++i) s += partials2[(size_t)i * 512 + j];
  tot[j] = s;
  __syncthreads();
  if (j < 256){
    float mean = tot[j] * invN;
    float var  = tot[j + 256] * invN - mean * mean;
    var = var < 0.f ? 0.f : var;
    float scale = gamma[j] * rsqrtf(var + 1e-5f);
    bn[j]       = scale;
    bn[256 + j] = beta[j] - mean * scale;
  }
}

// ---------------- stage2: BN, second Wc, attention, GRU fusion, final Wf ------------------
__global__ __launch_bounds__(32) void stage2(const float* __restrict__ g_h2,
                                             const float* __restrict__ bc,
                                             const float* __restrict__ wd,
                                             const float* __restrict__ bd,
                                             const float* __restrict__ bn,
                                             const float* __restrict__ bf,
                                             const h16*  __restrict__ wf,
                                             float* __restrict__ out){
  __shared__ h16 s0 [16 * HID];   // h2 -> q -> o -> rp
  __shared__ h16 s1 [16 * HID];   // bn_x -> k -> attn
  __shared__ h16 s2s[16 * HID];   // hc -> fin
  __shared__ h16 s3 [16 * HID];   // v -> z
  __shared__ float sc[16 * 64];   // softmax scores, per row 8x8
  int lane = threadIdx.x;
  long long row0 = (long long)blockIdx.x * 16;
  int cL = lane & 15;
  int rb = (lane & 16) ? 8 : 0;

  // reload h2 (f32 ws) -> s0 f16
  for (int idx = lane; idx < 16 * 64; idx += 32){
    int r = idx >> 6, q4 = idx & 63;
    float4 f = ((const float4*)(g_h2 + (row0 + r) * HID))[q4];
    h16* p = s0 + r * HID + q4 * 4;
    p[0]=(h16)f.x; p[1]=(h16)f.y; p[2]=(h16)f.z; p[3]=(h16)f.w;
  }
  __syncthreads();

  // step2: x = relu(BN(hc_pre)) with hc_pre = (h2@Wc+bc)*sig(h2)*wd+bd  -> s1
  {
    v16h A[8]; load_A<8>(A, s0, HID, lane);
    for (int cb = 0; cb < 16; ++cb){
      v8f acc = gemm_cb<8>(A, wf + OFF_WC, cb, lane, zero8());
      int c = cb * 16 + cL;
      float bb = bc[c], w = wd[c], bdd = bd[c], scl = bn[c], sh = bn[256 + c];
      #pragma unroll
      for (int v = 0; v < 8; ++v){
        int r = rb + v;
        float x = (acc[v] + bb) * sigm((float)s0[r * HID + c]);
        x = x * w + bdd;
        x = x * scl + sh;
        x = x > 0.f ? x : 0.f;
        s1[r * HID + c] = (h16)x;
      }
    }
  }
  __syncthreads();

  // step3: hc = x@Wc + bc + re(h2)  -> s2s
  {
    v16h A[8]; load_A<8>(A, s1, HID, lane);
    for (int cb = 0; cb < 16; ++cb){
      v8f acc = gemm_cb<8>(A, wf + OFF_WC, cb, lane, zero8());
      int c = cb * 16 + cL;
      float bb = bc[c];
      #pragma unroll
      for (int v = 0; v < 8; ++v){
        int r = rb + v;
        s2s[r * HID + c] = (h16)(acc[v] + bb + (float)s0[r * HID + c]);
      }
    }
  }
  __syncthreads();

  v16h Ahc[8]; load_A<8>(Ahc, s2s, HID, lane);   // hc fragments reused 6x

  // step4: Q -> s0, K -> s1, V -> s3 (no biases)
  for (int cb = 0; cb < 16; ++cb){
    int c = cb * 16 + cL;
    v8f aq = gemm_cb<8>(Ahc, wf + OFF_WQ, cb, lane, zero8());
    v8f ak = gemm_cb<8>(Ahc, wf + OFF_WK, cb, lane, zero8());
    v8f av = gemm_cb<8>(Ahc, wf + OFF_WV, cb, lane, zero8());
    #pragma unroll
    for (int v = 0; v < 8; ++v){
      int r = rb + v;
      s0[r * HID + c] = (h16)aq[v];
      s1[r * HID + c] = (h16)ak[v];
      s3[r * HID + c] = (h16)av[v];
    }
  }
  __syncthreads();

  // step5: per-row head-mixing attention (8x8 scores), scalar VALU — 2 lanes per row
  {
    int row = lane >> 1;
    int hi2 = lane & 1;
    const h16* qs = s0 + row * HID;
    const h16* ks = s1 + row * HID;
    #pragma unroll
    for (int hh = 0; hh < 4; ++hh){
      int a = hi2 * 4 + hh;              // comp row (Q head index)
      float cmp[8];
      #pragma unroll
      for (int b = 0; b < 8; ++b){
        float d = 0.f;
        #pragma unroll
        for (int k = 0; k < 32; ++k)
          d += (float)qs[a * 32 + k] * (float)ks[k * 8 + b];
        cmp[b] = d * (0.5f / 32.0f);
      }
      float m = cmp[0];
      #pragma unroll
      for (int b = 1; b < 8; ++b) m = cmp[b] > m ? cmp[b] : m;
      float sum = 0.f;
      #pragma unroll
      for (int b = 0; b < 8; ++b){ cmp[b] = __expf(cmp[b] - m); sum += cmp[b]; }
      float inv = 1.0f / sum;
      #pragma unroll
      for (int b = 0; b < 8; ++b) sc[row * 64 + a * 8 + b] = cmp[b] * inv;
    }
    __syncthreads();
    // out_flat[d*8 + a] = sum_b scores[a,b] * v[d*8 + b]  -> s0 (overwrites Q)
    const h16* vs = s3 + row * HID;
    h16* os = s0 + row * HID;
    int d0 = hi2 * 16;
    for (int d = d0; d < d0 + 16; ++d){
      #pragma unroll
      for (int a = 0; a < 8; ++a){
        float o = 0.f;
        #pragma unroll
        for (int b = 0; b < 8; ++b)
          o += sc[row * 64 + a * 8 + b] * (float)vs[d * 8 + b];
        os[d * 8 + a] = (h16)o;
      }
    }
  }
  __syncthreads();

  // step6: attn = o @ Wfc^T -> s1
  {
    v16h Ao[8]; load_A<8>(Ao, s0, HID, lane);
    for (int cb = 0; cb < 16; ++cb){
      v8f acc = gemm_cb<8>(Ao, wf + OFF_WFC, cb, lane, zero8());
      int c = cb * 16 + cL;
      #pragma unroll
      for (int v = 0; v < 8; ++v) s1[(rb + v) * HID + c] = (h16)acc[v];
    }
  }
  __syncthreads();

  v16h Aat[8]; load_A<8>(Aat, s1, HID, lane);    // attn fragments

  // step7a: z = sigmoid(hc@Wz + attn@Uz) -> s3
  for (int cb = 0; cb < 16; ++cb){
    v8f acc = gemm_cb<8>(Ahc, wf + OFF_WZ, cb, lane, zero8());
    acc = gemm_cb<8>(Aat, wf + OFF_UZ, cb, lane, acc);
    int c = cb * 16 + cL;
    #pragma unroll
    for (int v = 0; v < 8; ++v) s3[(rb + v) * HID + c] = (h16)sigm(acc[v]);
  }
  __syncthreads();

  // step7b: rp = sigmoid(hc@Wr + attn@Ur) * attn -> s0
  for (int cb = 0; cb < 16; ++cb){
    v8f acc = gemm_cb<8>(Ahc, wf + OFF_WR, cb, lane, zero8());
    acc = gemm_cb<8>(Aat, wf + OFF_UR, cb, lane, acc);
    int c = cb * 16 + cL;
    #pragma unroll
    for (int v = 0; v < 8; ++v){
      int r = rb + v;
      s0[r * HID + c] = (h16)(sigm(acc[v]) * (float)s1[r * HID + c]);
    }
  }
  __syncthreads();

  // step8: h_hat = tanh(hc@Wg + rp@Ug); out_h = ((1-z)*attn + z*h_hat)*sigmoid(attn) -> s2s
  {
    v16h Arp[8]; load_A<8>(Arp, s0, HID, lane);
    for (int cb = 0; cb < 16; ++cb){
      v8f acc = gemm_cb<8>(Ahc, wf + OFF_WG, cb, lane, zero8());
      acc = gemm_cb<8>(Arp, wf + OFF_UG, cb, lane, acc);
      int c = cb * 16 + cL;
      #pragma unroll
      for (int v = 0; v < 8; ++v){
        int r = rb + v;
        float hh = tanhf(acc[v]);
        float z  = (float)s3[r * HID + c];
        float at = (float)s1[r * HID + c];
        float oh = (1.0f - z) * at + z * hh;
        oh *= sigm(at);
        s2s[r * HID + c] = (h16)oh;
      }
    }
  }
  __syncthreads();

  // step9: out = fin @ Wf^T + bf  (f32 to global)
  {
    v16h Af[8]; load_A<8>(Af, s2s, HID, lane);
    for (int cb = 0; cb < 16; ++cb){
      v8f acc = gemm_cb<8>(Af, wf + OFF_WF, cb, lane, zero8());
      int c = cb * 16 + cL;
      float bb = bf[c];
      #pragma unroll
      for (int v = 0; v < 8; ++v)
        out[(row0 + rb + v) * HID + c] = acc[v] + bb;
    }
  }
}

// ---------------- host launch --------------------------------------------------------------
extern "C" void kernel_launch(void* const* d_in, const int* in_sizes, int n_in,
                              void* d_out, int out_size, void* d_ws, size_t ws_size,
                              hipStream_t stream) {
  const float* h     = (const float*)d_in[0];
  const float* b1    = (const float*)d_in[2];
  const float* b2    = (const float*)d_in[4];
  const float* bc    = (const float*)d_in[6];
  const float* wd    = (const float*)d_in[7];
  const float* bd    = (const float*)d_in[8];
  const float* gamma = (const float*)d_in[9];
  const float* beta  = (const float*)d_in[10];
  const float* bf    = (const float*)d_in[22];

  int N  = in_sizes[0] / HID;   // 200000
  int NB = N / 16;              // 12500 row-tiles (N divisible by 16 here)

  char* ws = (char*)d_ws;
  h16*   wf       = (h16*)ws;
  float* g_h2     = (float*)(ws + W_HALVES * 2);
  size_t pOff     = W_HALVES * 2 + (size_t)N * HID * 4;
  float* partials = (float*)(ws + pOff);
  const int CH = 100;
  int NB1 = (NB + CH - 1) / CH;
  size_t p2Off    = pOff + (size_t)NB * 512 * 4;
  float* partials2= (float*)(ws + p2Off);
  float* bnbuf    = (float*)(ws + p2Off + (size_t)NB1 * 512 * 4);

  struct M { const float* src; size_t off; int O, K; };
  M mats[14] = {
    { (const float*)d_in[1],  OFF_W1,  FFNW, HID  },
    { (const float*)d_in[3],  OFF_W2,  HID,  FFNW },
    { (const float*)d_in[5],  OFF_WC,  HID,  HID  },
    { (const float*)d_in[11], OFF_WQ,  HID,  HID  },
    { (const float*)d_in[12], OFF_WK,  HID,  HID  },
    { (const float*)d_in[13], OFF_WV,  HID,  HID  },
    { (const float*)d_in[14], OFF_WFC, HID,  HID  },
    { (const float*)d_in[15], OFF_WZ,  HID,  HID  },
    { (const float*)d_in[16], OFF_UZ,  HID,  HID  },
    { (const float*)d_in[17], OFF_WR,  HID,  HID  },
    { (const float*)d_in[18], OFF_UR,  HID,  HID  },
    { (const float*)d_in[19], OFF_WG,  HID,  HID  },
    { (const float*)d_in[20], OFF_UG,  HID,  HID  },
    { (const float*)d_in[21], OFF_WF,  HID,  HID  },
  };
  for (int i = 0; i < 14; ++i){
    int frags = (mats[i].O / 16) * (mats[i].K / 32);
    prep_w<<<frags, 32, 0, stream>>>(mats[i].src, wf + mats[i].off, mats[i].O, mats[i].K);
  }

  stage1<<<NB, 32, 0, stream>>>(h, b1, b2, bc, wd, bd, wf, g_h2, partials);
  bn_reduce1<<<NB1, 512, 0, stream>>>(partials, partials2, NB, CH);
  bn_reduce2<<<1, 512, 0, stream>>>(partials2, gamma, beta, bnbuf, NB1, 1.0f / (float)N);
  stage2<<<NB, 32, 0, stream>>>(g_h2, bc, wd, bd, bnbuf, bf, wf, (float*)d_out);
}
